// MultiHeadAttention_21002390077931
// MI455X (gfx1250) — matrix operand order repaired and checked by reading, not compiled, not run on hardware
//
#include <hip/hip_runtime.h>
#include <stdint.h>

// ---------------------------------------------------------------------------
// MHA forward for MI455X (gfx1250, wave32, WMMA + TDM).
// Pipeline: pack(x,W*) -> QKV GEMM (bf16 WMMA) -> flash attention (TDM-staged
// K/V tiles in LDS, bf16 WMMA, online softmax, causal) -> output GEMM.
// Workspace layout (bytes):
//   [0,8M)    xbf   : x as bf16, (B*S, DM) row-major
//   [8M,10M)  WqT   : Wq^T bf16 (N=1024, K=1024)
//   [10M,12M) WkT
//   [12M,14M) WvT
//   [14M,16M) WoT
//   [16M,24M) Qb    : (B,H,S,D) bf16
//   [24M,32M) Kb    : (B,H,S,D) bf16
//   [32M,40M) Vt    : (B,H,D,S) bf16 (transposed for PV B-fragments)
//   [40M,48M) attn  : (B,S,H*D) bf16
// ---------------------------------------------------------------------------

typedef __attribute__((ext_vector_type(16))) __bf16    v16bf;
typedef __attribute__((ext_vector_type(8)))  float     v8f;
typedef __attribute__((ext_vector_type(4)))  uint32_t  u32x4;
typedef __attribute__((ext_vector_type(8)))  int32_t   i32x8;
typedef __attribute__((ext_vector_type(4)))  int32_t   i32x4;

#define B_   2
#define S_   2048
#define DM_  1024
#define H_   16
#define D_   64
#define BH_  (B_ * H_)
#define M_   (B_ * S_)      // 4096 token rows

union FragBF {
  v16bf bf;
  u32x4 u[2];
};

__device__ __forceinline__ uint16_t f2bf(float f) {
  union { float f; uint32_t u; } c; c.f = f;
  uint32_t u = c.u;
  return (uint16_t)((u + 0x7FFFu + ((u >> 16) & 1u)) >> 16);  // RNE
}

// --------------------- Tensor Data Mover (2-D tile load) -------------------
// D# per CDNA5 ISA ch.8: group0 {count=1, lds_addr, global_addr, type=2},
// group1 {data_size=1 (2B), tensor_dim0/1 == tile dims (exact fit -> no OOB),
// tile_dim0/1, tensor_dim0_stride}. Groups 2/3 zero (2-D tensor).
__device__ __forceinline__ void tdm_load_2d(uint32_t lds_off, const void* gptr,
                                            uint32_t tile_d0, uint32_t tile_d1,
                                            uint32_t stride0_elems) {
  const uint64_t ga = (uint64_t)(uintptr_t)gptr;
  u32x4 g0;
  g0[0] = 1u;                                   // count=1 (valid user D#)
  g0[1] = lds_off;                              // lds_addr (bytes)
  g0[2] = (uint32_t)ga;                         // global_addr[31:0]
  g0[3] = (uint32_t)(ga >> 32) | (2u << 30);    // global_addr[56:32] | type=2
  i32x8 g1;
  g1[0] = (int32_t)(1u << 16);                  // workgroup_mask=0, data_size=1 (2B)
  g1[1] = (int32_t)(tile_d0 << 16);             // [63:48] tensor_dim0[15:0]
  g1[2] = (int32_t)((tile_d1 & 0xFFFFu) << 16); // [95:80] tensor_dim1[15:0]
  g1[3] = (int32_t)(tile_d0 << 16);             // [127:112] tile_dim0
  g1[4] = (int32_t)(tile_d1 & 0xFFFFu);         // [143:128] tile_dim1, tile_dim2=0
  g1[5] = (int32_t)stride0_elems;               // tensor_dim0_stride[31:0]
  g1[6] = 0;
  g1[7] = 0;
  const i32x4 z4 = {0, 0, 0, 0};
#if defined(__clang_major__) && __clang_major__ >= 23
  const i32x8 z8 = {0, 0, 0, 0, 0, 0, 0, 0};
  __builtin_amdgcn_tensor_load_to_lds(g0, g1, z4, z4, z8, 0);
#else
  __builtin_amdgcn_tensor_load_to_lds(g0, g1, z4, z4, 0);
#endif
}

// ------------------------------- pack kernels ------------------------------

__global__ void pack_x_kernel(const float* __restrict__ x,
                              uint16_t* __restrict__ xbf, int n) {
  int i = blockIdx.x * blockDim.x + threadIdx.x;
  if (i < n) xbf[i] = f2bf(x[i]);
}

// W: (K=1024, N=1024) row-major  ->  WT: (N, K) bf16
__global__ void pack_wT_kernel(const float* __restrict__ W,
                               uint16_t* __restrict__ WT) {
  int i = blockIdx.x * blockDim.x + threadIdx.x;   // 1M threads
  int k = i >> 10;
  int n = i & 1023;                                // consecutive -> coalesced read
  WT[(size_t)n * 1024 + k] = f2bf(W[(size_t)k * 1024 + n]);
}

// ------------------------- QKV projection GEMM -----------------------------
// grid = (M/128, 1024/64, 3), block = 256 (8 waves). Wave owns 16 rows x 64 cols.

__global__ void gemm_qkv_kernel(const uint16_t* __restrict__ xbf,
                                const uint16_t* __restrict__ WqT,
                                const uint16_t* __restrict__ WkT,
                                const uint16_t* __restrict__ WvT,
                                const float* __restrict__ bq,
                                const float* __restrict__ bk,
                                const float* __restrict__ bv,
                                uint16_t* __restrict__ Qo,
                                uint16_t* __restrict__ Ko,
                                uint16_t* __restrict__ Vt) {
  const int wave = threadIdx.x >> 5, lane = threadIdx.x & 31;
  const int half = lane >> 4, lm = lane & 15;
  const int m0 = blockIdx.x * 128 + wave * 16;
  const int n0 = blockIdx.y * 64;
  const int z  = blockIdx.z;

  const uint16_t* WT  = (z == 0) ? WqT : (z == 1) ? WkT : WvT;
  const float*   bias = (z == 0) ? bq  : (z == 1) ? bk  : bv;

  v8f acc[4];
#pragma unroll
  for (int t = 0; t < 4; ++t)
#pragma unroll
    for (int v = 0; v < 8; ++v) acc[t][v] = 0.0f;

  const uint16_t* arow = xbf + (size_t)(m0 + lm) * DM_;

  for (int k0 = 0; k0 < DM_; k0 += 32) {
    FragBF a;  // A 16x32: lanes 0-15 row=lm K=[k0..k0+7 | k0+16..23]; lanes 16-31 +8
    a.u[0] = *(const u32x4*)(arow + k0 + half * 8);
    a.u[1] = *(const u32x4*)(arow + k0 + half * 8 + 16);
#pragma unroll
    for (int t = 0; t < 4; ++t) {
      const uint16_t* brow = WT + (size_t)(n0 + t * 16 + lm) * DM_ + k0 + half * 8;
      FragBF b;  // B 32x16: column N=lm, K=[k0+8h..+7 | k0+16+8h..+7]
      b.u[0] = *(const u32x4*)(brow);
      b.u[1] = *(const u32x4*)(brow + 16);
      acc[t] = __builtin_amdgcn_wmma_f32_16x16x32_bf16(
          false, a.bf, false, b.bf, (short)0, acc[t], false, false);
    }
  }

#pragma unroll
  for (int t = 0; t < 4; ++t) {
    const int n = n0 + t * 16 + lm;
    const int h = n >> 6, d = n & 63;
    const float bv_ = bias[n];
#pragma unroll
    for (int v = 0; v < 8; ++v) {
      const int m = m0 + v + 8 * half;        // C tile: M = v + 8*(lane>=16)
      const int b = m >> 11, s = m & (S_ - 1);
      const uint16_t val = f2bf(acc[t][v] + bv_);
      if (z == 2) {
        Vt[((size_t)(b * H_ + h) * D_ + d) * S_ + s] = val;
      } else {
        uint16_t* out = (z == 0) ? Qo : Ko;
        out[((size_t)(b * H_ + h) * S_ + s) * D_ + d] = val;
      }
    }
  }
}

// ----------------------------- flash attention -----------------------------
// grid = (S/128, B*H), block = 256 (8 waves, each 16 query rows).
// K/V tiles staged into LDS by the Tensor Data Mover (double-buffered),
// fragments read via ds_load_b128, matmuls via v_wmma_f32_16x16x32_bf16.

__global__ void flash_attn_kernel(const uint16_t* __restrict__ Qb,
                                  const uint16_t* __restrict__ Kb,
                                  const uint16_t* __restrict__ Vt,
                                  uint16_t* __restrict__ attn) {
  __shared__ __attribute__((aligned(16))) uint16_t Ktile[2][32 * 64]; // [s][d]
  __shared__ __attribute__((aligned(16))) uint16_t Vtile[2][64 * 32]; // [d][s]
  __shared__ __attribute__((aligned(16))) uint16_t Pb[8][16 * 48];    // per-wave P

  const int wave = threadIdx.x >> 5, lane = threadIdx.x & 31;
  const int half = lane >> 4, lm = lane & 15;
  const int bh   = blockIdx.y;
  const int qrow = blockIdx.x * 128 + wave * 16;

  // Q fragments for the 16-row strip (K-dim = D = 64 -> two 16x32 A fragments)
  const uint16_t* Qrow = Qb + ((size_t)bh * S_ + qrow + lm) * D_;
  FragBF qa[2];
#pragma unroll
  for (int j = 0; j < 2; ++j) {
    qa[j].u[0] = *(const u32x4*)(Qrow + 32 * j + half * 8);
    qa[j].u[1] = *(const u32x4*)(Qrow + 32 * j + half * 8 + 16);
  }

  v8f o[4];
  float mi[8], li[8];
#pragma unroll
  for (int t = 0; t < 4; ++t)
#pragma unroll
    for (int v = 0; v < 8; ++v) o[t][v] = 0.0f;
#pragma unroll
  for (int v = 0; v < 8; ++v) { mi[v] = -1e30f; li[v] = 0.0f; }

  const int nkt = blockIdx.x * 4 + 4;   // causal bound, uniform per block
  const float scale = 0.125f;           // 1/sqrt(64)
  uint16_t* pbw = &Pb[wave][0];

  const uint16_t* Kbase = Kb + (size_t)bh * S_ * D_;
  const uint16_t* Vbase = Vt + (size_t)bh * D_ * S_;

  auto issue_tile = [&](int kt2, int buf) {
    // K tile: 32 key rows x 64 d, row-major in Kb -> contiguous rows.
    tdm_load_2d((uint32_t)(uintptr_t)&Ktile[buf][0],
                Kbase + (size_t)kt2 * 32 * D_, /*d0=*/64, /*d1=*/32, /*stride=*/D_);
    // V tile: 64 d rows x 32 keys from transposed Vt (row stride S).
    tdm_load_2d((uint32_t)(uintptr_t)&Vtile[buf][0],
                Vbase + (size_t)kt2 * 32, /*d0=*/32, /*d1=*/64, /*stride=*/S_);
  };

  if (wave == 0) issue_tile(0, 0);      // prologue: tile 0 in flight

  for (int kt = 0; kt < nkt; ++kt) {
    const int k0  = kt * 32;
    const int buf = kt & 1;

    if (wave == 0) {
      if (kt + 1 < nkt) {
        issue_tile(kt + 1, buf ^ 1);              // prefetch next tile
        __builtin_amdgcn_s_wait_tensorcnt(2);     // TDM in-order: current tile done
      } else {
        __builtin_amdgcn_s_wait_tensorcnt(0);
      }
    }
    __syncthreads();                              // tile visible to all waves

    // ---- scores: two 16x16 tiles (k columns k0..k0+31), K-dim = D = 64 ----
    v8f sc[2];
#pragma unroll
    for (int n = 0; n < 2; ++n) {
#pragma unroll
      for (int v = 0; v < 8; ++v) sc[n][v] = 0.0f;
      const uint16_t* Krow = &Ktile[buf][(n * 16 + lm) * D_];
#pragma unroll
      for (int j = 0; j < 2; ++j) {
        FragBF kf;  // B frag: column N=lm is key row, d=[8h..+7 | 16+8h..+7]
        kf.u[0] = *(const u32x4*)(Krow + 32 * j + half * 8);
        kf.u[1] = *(const u32x4*)(Krow + 32 * j + half * 8 + 16);
        sc[n] = __builtin_amdgcn_wmma_f32_16x16x32_bf16(
            false, qa[j].bf, false, kf.bf, (short)0, sc[n], false, false);
      }
    }

    // ---- scale + causal mask + online softmax (per row = per (v, half)) ----
    float p0a[8], p1a[8];
#pragma unroll
    for (int v = 0; v < 8; ++v) {
      const int qg = qrow + v + 8 * half;
      float s0 = sc[0][v] * scale;
      float s1 = sc[1][v] * scale;
      if (k0 + lm      > qg) s0 = -1e30f;
      if (k0 + 16 + lm > qg) s1 = -1e30f;
      float t = fmaxf(s0, s1);
#pragma unroll
      for (int off = 1; off < 16; off <<= 1) t = fmaxf(t, __shfl_xor(t, off, 32));
      const float mnew = fmaxf(mi[v], t);
      const float fscl = __expf(mi[v] - mnew);
      const float p0 = __expf(s0 - mnew);
      const float p1 = __expf(s1 - mnew);
      float r = p0 + p1;
#pragma unroll
      for (int off = 1; off < 16; off <<= 1) r += __shfl_xor(r, off, 32);
      li[v] = li[v] * fscl + r;
      mi[v] = mnew;
#pragma unroll
      for (int t4 = 0; t4 < 4; ++t4) o[t4][v] *= fscl;
      p0a[v] = p0; p1a[v] = p1;
    }

    // ---- P (C layout) -> LDS -> A fragment layout (per-wave, LDS in-order) ----
#pragma unroll
    for (int v = 0; v < 8; ++v) {
      const int row = v + 8 * half;
      pbw[row * 48 + lm]      = f2bf(p0a[v]);
      pbw[row * 48 + 16 + lm] = f2bf(p1a[v]);
    }
    __asm__ volatile("s_wait_dscnt 0x0" ::: "memory");
    FragBF pa;
    pa.u[0] = *(const u32x4*)(pbw + lm * 48 + half * 8);
    pa.u[1] = *(const u32x4*)(pbw + lm * 48 + half * 8 + 16);

    // ---- O += P @ V from LDS V tile (d-major: contiguous key-range per lane) ----
#pragma unroll
    for (int t4 = 0; t4 < 4; ++t4) {
      const uint16_t* Vrow = &Vtile[buf][(t4 * 16 + lm) * 32 + half * 8];
      FragBF vf;
      vf.u[0] = *(const u32x4*)(Vrow);
      vf.u[1] = *(const u32x4*)(Vrow + 16);
      o[t4] = __builtin_amdgcn_wmma_f32_16x16x32_bf16(
          false, pa.bf, false, vf.bf, (short)0, o[t4], false, false);
    }
    __syncthreads();   // protect buf^1... buf reuse two iterations ahead
  }

  // ---- finalize: divide by row sums, store (b, s, h*64+d) bf16 ----
  const int b = bh >> 4, h = bh & 15;
#pragma unroll
  for (int t4 = 0; t4 < 4; ++t4)
#pragma unroll
    for (int v = 0; v < 8; ++v) {
      const int s = qrow + v + 8 * half;
      const float val = o[t4][v] / li[v];
      attn[((size_t)b * S_ + s) * (H_ * D_) + h * D_ + t4 * 16 + lm] = f2bf(val);
    }
}

// --------------------------- output projection -----------------------------
// grid = (M/128, 1024/64), block = 256. Same tiling; f32 output + bias.

__global__ void gemm_out_kernel(const uint16_t* __restrict__ attn,
                                const uint16_t* __restrict__ WoT,
                                const float* __restrict__ bo,
                                float* __restrict__ out) {
  const int wave = threadIdx.x >> 5, lane = threadIdx.x & 31;
  const int half = lane >> 4, lm = lane & 15;
  const int m0 = blockIdx.x * 128 + wave * 16;
  const int n0 = blockIdx.y * 64;

  v8f acc[4];
#pragma unroll
  for (int t = 0; t < 4; ++t)
#pragma unroll
    for (int v = 0; v < 8; ++v) acc[t][v] = 0.0f;

  const uint16_t* arow = attn + (size_t)(m0 + lm) * DM_;

  for (int k0 = 0; k0 < DM_; k0 += 32) {
    FragBF a;
    a.u[0] = *(const u32x4*)(arow + k0 + half * 8);
    a.u[1] = *(const u32x4*)(arow + k0 + half * 8 + 16);
#pragma unroll
    for (int t = 0; t < 4; ++t) {
      const uint16_t* brow = WoT + (size_t)(n0 + t * 16 + lm) * DM_ + k0 + half * 8;
      FragBF bfr;
      bfr.u[0] = *(const u32x4*)(brow);
      bfr.u[1] = *(const u32x4*)(brow + 16);
      acc[t] = __builtin_amdgcn_wmma_f32_16x16x32_bf16(
          false, a.bf, false, bfr.bf, (short)0, acc[t], false, false);
    }
  }

#pragma unroll
  for (int t = 0; t < 4; ++t) {
    const int n = n0 + t * 16 + lm;
    const float bv_ = bo[n];
#pragma unroll
    for (int v = 0; v < 8; ++v) {
      const int m = m0 + v + 8 * half;
      out[(size_t)m * DM_ + n] = acc[t][v] + bv_;
    }
  }
}

// ------------------------------- launcher ----------------------------------

extern "C" void kernel_launch(void* const* d_in, const int* in_sizes, int n_in,
                              void* d_out, int out_size, void* d_ws, size_t ws_size,
                              hipStream_t stream) {
  const float* x  = (const float*)d_in[0];
  const float* Wq = (const float*)d_in[1];
  const float* bq = (const float*)d_in[2];
  const float* Wk = (const float*)d_in[3];
  const float* bk = (const float*)d_in[4];
  const float* Wv = (const float*)d_in[5];
  const float* bv = (const float*)d_in[6];
  const float* Wo = (const float*)d_in[7];
  const float* bo = (const float*)d_in[8];
  float* out = (float*)d_out;

  char* ws = (char*)d_ws;
  const size_t MB = 1024 * 1024;
  uint16_t* xbf  = (uint16_t*)(ws + 0);
  uint16_t* WqT  = (uint16_t*)(ws + 8  * MB);
  uint16_t* WkT  = (uint16_t*)(ws + 10 * MB);
  uint16_t* WvT  = (uint16_t*)(ws + 12 * MB);
  uint16_t* WoT  = (uint16_t*)(ws + 14 * MB);
  uint16_t* Qb   = (uint16_t*)(ws + 16 * MB);
  uint16_t* Kb   = (uint16_t*)(ws + 24 * MB);
  uint16_t* Vt   = (uint16_t*)(ws + 32 * MB);
  uint16_t* attn = (uint16_t*)(ws + 40 * MB);

  // 1) pack inputs to bf16 (weights transposed for contiguous B fragments)
  pack_x_kernel<<<(M_ * DM_) / 256, 256, 0, stream>>>(x, xbf, M_ * DM_);
  pack_wT_kernel<<<(DM_ * DM_) / 256, 256, 0, stream>>>(Wq, WqT);
  pack_wT_kernel<<<(DM_ * DM_) / 256, 256, 0, stream>>>(Wk, WkT);
  pack_wT_kernel<<<(DM_ * DM_) / 256, 256, 0, stream>>>(Wv, WvT);
  pack_wT_kernel<<<(DM_ * DM_) / 256, 256, 0, stream>>>(Wo, WoT);

  // 2) fused QKV projection (z picks Q/K/V; V stored transposed)
  gemm_qkv_kernel<<<dim3(M_ / 128, DM_ / 64, 3), 256, 0, stream>>>(
      xbf, WqT, WkT, WvT, bq, bk, bv, Qb, Kb, Vt);

  // 3) causal flash attention with TDM-staged K/V tiles
  flash_attn_kernel<<<dim3(S_ / 128, BH_), 256, 0, stream>>>(Qb, Kb, Vt, attn);

  // 4) output projection (f32 out + bias)
  gemm_out_kernel<<<dim3(M_ / 128, DM_ / 64), 256, 0, stream>>>(attn, WoT, bo, out);
}
